// GNNBase_67284957659199
// MI455X (gfx1250) — matrix operand order repaired and checked by reading, not compiled, not run on hardware
//
#include <hip/hip_runtime.h>
#include <hip/hip_bf16.h>
#include <math.h>

// ---------------------------------------------------------------------------
// GNN edge-attention pipeline for MI455X (gfx1250, wave32, WMMA f16 16x16x32)
//
// ~120 GFLOP of small-K GEMMs vs 2 MB of adj -> compute bound. Per-wave
// 16-edge tiles, edges on the N dimension (D = W^T * X), fused online softmax.
// All four big weight matrices are staged into LDS once per block with a
// single TDM tensor_load_to_lds (f16, bank-conflict-padded rows), so the
// steady state is WMMA + ds_load only.
// ---------------------------------------------------------------------------

typedef __attribute__((ext_vector_type(16))) _Float16 v16h;
typedef __attribute__((ext_vector_type(8)))  _Float16 v8h;
typedef __attribute__((ext_vector_type(8)))  float    v8f;
typedef __attribute__((ext_vector_type(4)))  unsigned int u32x4;
typedef __attribute__((ext_vector_type(8)))  unsigned int u32x8;

#define WPB 4  // waves per block (128 threads)

// Padded-row f16 weight arena (pad +8 halfs/row => lane stride = 4 banks).
//   W1bT [128][264] (k<256 valid)   @ 0
//   W2aT [128][136] (k<128 valid)   @ 33792
//   W3aT [256][136] (k<128 valid)   @ 51200
//   W3bT [128][264] (k<256 valid)   @ 86016
#define O_W1BT 0
#define O_W2AT 33792
#define O_W3AT 51200
#define O_W3BT 86016
#define W_TOTAL 119808           // halfs (= 239616 bytes = 59904 dwords)
#define S1 264                   // padded stride for k=256 rows
#define S2 136                   // padded stride for k=128 rows
#define EMB_S 136                // emb scratch row stride (halfs)
#define H3_S 264                 // h3 scratch row stride (halfs)

__device__ __forceinline__ v8f wmma_f32_16x16x32(v16h a, v16h b, v8f c) {
  return __builtin_amdgcn_wmma_f32_16x16x32_f16(false, a, false, b, (short)0, c,
                                                false, false);
}

// A-operand fragment (16x32 f16, this code's contiguous k order): lane = M (mod 16).
// lane<16 : elems 0..15 -> K 0..15
// lane>=16: elems 0..15 -> K 16..31
__device__ __forceinline__ v16h load_afrag(const _Float16* row, int kb,
                                           int half) {
  const _Float16* p0 = row + kb * 32 + (half ? 16 : 0);
  const _Float16* p1 = p0 + 8;
  v8h lo = *(const v8h*)p0;
  v8h hi = *(const v8h*)p1;
  v16h a;
#pragma unroll
  for (int i = 0; i < 8; ++i) { a[i] = lo[i]; a[i + 8] = hi[i]; }
  return a;
}

// B-operand fragment (32x16 f16): lane = N (mod 16); lane-half = K 0..15 vs
// 16..31; 16 contiguous K f16 per lane.
__device__ __forceinline__ v16h load_bfrag(const _Float16* p) {
  v8h lo = *(const v8h*)p;
  v8h hi = *(const v8h*)(p + 8);
  v16h b;
#pragma unroll
  for (int i = 0; i < 8; ++i) { b[i] = lo[i]; b[i + 8] = hi[i]; }
  return b;
}

// ---------------------------------------------------------------------------
// Kernel 1: build padded f16 transposed-weight arena in workspace.
// ---------------------------------------------------------------------------
__global__ void gnn_prep_kernel(const float* __restrict__ W1b,
                                const float* __restrict__ W2a,
                                const float* __restrict__ W3a,
                                const float* __restrict__ W3b,
                                _Float16* __restrict__ hw) {
  for (int i = blockIdx.x * blockDim.x + threadIdx.x; i < W_TOTAL;
       i += gridDim.x * blockDim.x) {
    float v = 0.f;
    if (i < O_W2AT) {                       // W1bT [m][k] from W1b[k*128+m]
      int m = i / S1, k = i % S1;
      if (k < 256) v = W1b[k * 128 + m];
    } else if (i < O_W3AT) {                // W2aT from W2a[k*128+m]
      int j = i - O_W2AT, m = j / S2, k = j % S2;
      if (k < 128) v = W2a[k * 128 + m];
    } else if (i < O_W3BT) {                // W3aT from W3a[k*256+m]
      int j = i - O_W3AT, m = j / S2, k = j % S2;
      if (k < 128) v = W3a[k * 256 + m];
    } else {                                // W3bT from W3b[k*128+m]
      int j = i - O_W3BT, m = j / S1, k = j % S1;
      if (k < 256) v = W3b[k * 128 + m];
    }
    hw[i] = (_Float16)v;
  }
}

// ---------------------------------------------------------------------------
// Kernel 2: per-edge pipeline + fused online softmax accumulation.
// ---------------------------------------------------------------------------
__global__ __launch_bounds__(WPB * 32) void gnn_edge_kernel(
    const float* __restrict__ adj, int P, int nTiles, int nWaves,
    const _Float16* __restrict__ hw,
    const float* __restrict__ W1a, const float* __restrict__ b1a,
    const float* __restrict__ b1b, const float* __restrict__ b2a,
    const float* __restrict__ W2b, const float* __restrict__ b3a,
    const float* __restrict__ b3b, float* __restrict__ partials) {
  __shared__ __align__(16) _Float16 s_w[W_TOTAL];          // 239616 B
  __shared__ __align__(16) _Float16 s_emb[WPB][16 * EMB_S]; // 17408 B
  __shared__ __align__(16) _Float16 s_h3[WPB][16 * H3_S];   // 33792 B
  __shared__ float s_W1a[256], s_b1a[256], s_b3a[256];
  __shared__ float s_b1b[128], s_b2a[128], s_W2b[128], s_b3b[128];

  const int tid = threadIdx.x;
  for (int i = tid; i < 256; i += WPB * 32) {
    s_W1a[i] = W1a[i]; s_b1a[i] = b1a[i]; s_b3a[i] = b3a[i];
  }
  for (int i = tid; i < 128; i += WPB * 32) {
    s_b1b[i] = b1b[i]; s_b2a[i] = b2a[i]; s_W2b[i] = W2b[i]; s_b3b[i] = b3b[i];
  }

  // ---- TDM: stage the whole f16 weight arena into LDS (wave 0 only) ------
  if (tid < 32) {
    unsigned long long ga = (unsigned long long)(size_t)hw;
    unsigned lds_off = (unsigned)(size_t)(&s_w[0]);
    const unsigned ND = W_TOTAL * 2 / 4;  // 59904 dwords
    u32x4 g0;
    g0[0] = 1u;                                        // count=1
    g0[1] = lds_off;                                   // lds_addr
    g0[2] = (unsigned)(ga & 0xFFFFFFFFu);              // global_addr[31:0]
    g0[3] = ((unsigned)(ga >> 32) & 0x1FFFFFFu) | (2u << 30);  // type=2
    u32x8 g1;
    g1[0] = 2u << 16;                 // data_size = 4 B
    g1[1] = (ND & 0xFFFFu) << 16;     // tensor_dim0[15:0]
    g1[2] = (ND >> 16) | (1u << 16);  // tensor_dim0[31:16], tensor_dim1=1
    g1[3] = (ND & 0xFFFFu) << 16;     // tile_dim0 = ND
    g1[4] = 1u;                       // tile_dim1 = 1, tile_dim2 = 0
    g1[5] = ND;                       // tensor_dim0_stride[31:0]
    g1[6] = (ND & 0xFFFFu) << 16;     // dim0_stride[47:32]=0, dim1_stride lo
    g1[7] = 0u;
    asm volatile("tensor_load_to_lds %0, %1" ::"s"(g0), "s"(g1) : "memory");
    __builtin_amdgcn_s_wait_tensorcnt(0);
  }
  __syncthreads();

  const int lane  = tid & 31;
  const int wslot = tid >> 5;
  const int wid   = (blockIdx.x * (WPB * 32) + tid) >> 5;
  const int half  = lane >> 4;
  const int n     = lane & 15;
  const float NEGINF = -__builtin_inff();

  const _Float16* w1bt = s_w + O_W1BT;
  const _Float16* w2at = s_w + O_W2AT;
  const _Float16* w3at = s_w + O_W3AT;
  const _Float16* w3bt = s_w + O_W3BT;
  _Float16* embL = s_emb[wslot];
  _Float16* h3L  = s_h3[wslot];

  float mx = NEGINF, Zp = 0.f;
  float acc[8][8];
#pragma unroll
  for (int a = 0; a < 8; ++a)
#pragma unroll
    for (int b = 0; b < 8; ++b) acc[a][b] = 0.f;

  for (int t = wid; t < nTiles; t += nWaves) {
    const int p = t * 16 + n;
    float av = (p < P) ? adj[p] : 0.f;
    const bool valid = (p < P) && (av > 0.f) && (av < 1.0f);
    const float e = valid ? av : 0.f;

    // ---- stage 2: emb[128 x 16] = W1b^T * relu(e*W1a + b1a) --------------
    v8f embD[8];
#pragma unroll
    for (int mt = 0; mt < 8; ++mt)
#pragma unroll
      for (int r = 0; r < 8; ++r) embD[mt][r] = 0.f;

#pragma unroll
    for (int kb = 0; kb < 8; ++kb) {
      v16h bf;
      const int kbase = kb * 32 + 16 * half;
#pragma unroll
      for (int j = 0; j < 16; ++j) {
        float h = fmaf(e, s_W1a[kbase + j], s_b1a[kbase + j]);
        bf[j] = (_Float16)fmaxf(h, 0.f);
      }
#pragma unroll
      for (int mt = 0; mt < 8; ++mt) {
        v16h af = load_afrag(w1bt + (mt * 16 + n) * S1, kb, half);
        embD[mt] = wmma_f32_16x16x32(af, bf, embD[mt]);
      }
    }
#pragma unroll
    for (int mt = 0; mt < 8; ++mt) {
      const int m0 = mt * 16 + 8 * half;
#pragma unroll
      for (int r = 0; r < 8; ++r)
        embL[n * EMB_S + m0 + r] = (_Float16)(embD[mt][r] + s_b1b[m0 + r]);
    }
    asm volatile("s_wait_dscnt 0" ::: "memory");

    v16h ef[4];  // emb as B-operand, reused by stages 3 and 4
#pragma unroll
    for (int kb = 0; kb < 4; ++kb)
      ef[kb] = load_bfrag(&embL[n * EMB_S + kb * 32 + 16 * half]);

    // ---- stage 3: logits = relu(emb @ W2a + b2a) @ W2b -------------------
    float plog = 0.f;
#pragma unroll
    for (int mt = 0; mt < 8; ++mt) {
      v8f h2;
#pragma unroll
      for (int r = 0; r < 8; ++r) h2[r] = 0.f;
#pragma unroll
      for (int kb = 0; kb < 4; ++kb) {
        v16h af = load_afrag(w2at + (mt * 16 + n) * S2, kb, half);
        h2 = wmma_f32_16x16x32(af, ef[kb], h2);
      }
      const int m0 = mt * 16 + 8 * half;
#pragma unroll
      for (int r = 0; r < 8; ++r) {
        float v = fmaxf(h2[r] + s_b2a[m0 + r], 0.f);
        plog = fmaf(v, s_W2b[m0 + r], plog);
      }
    }
    float lg = plog + __shfl_xor(plog, 16, 32);   // join lane-halves
    // (b2b is wave-uniform -> cancels under softmax; skipped exactly)
    float lgv = valid ? lg : NEGINF;
    float tm = lgv;
    tm = fmaxf(tm, __shfl_xor(tm, 1, 32));
    tm = fmaxf(tm, __shfl_xor(tm, 2, 32));
    tm = fmaxf(tm, __shfl_xor(tm, 4, 32));
    tm = fmaxf(tm, __shfl_xor(tm, 8, 32));
    float nmx   = fmaxf(mx, tm);
    float scale = (nmx == NEGINF) ? 1.f : __expf(mx - nmx);
    float w     = valid ? __expf(lg - nmx) : 0.f;
    mx = nmx;
    Zp = Zp * scale + w;

    // ---- stage 4: h3[256 x 16] = relu(W3a^T * emb + b3a) -----------------
#pragma unroll
    for (int mt = 0; mt < 16; ++mt) {
      v8f d;
#pragma unroll
      for (int r = 0; r < 8; ++r) d[r] = 0.f;
#pragma unroll
      for (int kb = 0; kb < 4; ++kb) {
        v16h af = load_afrag(w3at + (mt * 16 + n) * S2, kb, half);
        d = wmma_f32_16x16x32(af, ef[kb], d);
      }
      const int m0 = mt * 16 + 8 * half;
#pragma unroll
      for (int r = 0; r < 8; ++r)
        h3L[n * H3_S + m0 + r] = (_Float16)fmaxf(d[r] + s_b3a[m0 + r], 0.f);
    }
    asm volatile("s_wait_dscnt 0" ::: "memory");

    // ---- stage 5: vals[128 x 16] = W3b^T * h3 + b3b; acc += w*vals -------
#pragma unroll
    for (int mt = 0; mt < 8; ++mt) {
      v8f d;
#pragma unroll
      for (int r = 0; r < 8; ++r) d[r] = 0.f;
#pragma unroll
      for (int kb = 0; kb < 8; ++kb) {
        v16h af = load_afrag(w3bt + (mt * 16 + n) * S1, kb, half);
        v16h bf = load_bfrag(&h3L[n * H3_S + kb * 32 + 16 * half]);
        d = wmma_f32_16x16x32(af, bf, d);
      }
      const int m0 = mt * 16 + 8 * half;
#pragma unroll
      for (int r = 0; r < 8; ++r) {
        float v = d[r] + s_b3b[m0 + r];
        acc[mt][r] = acc[mt][r] * scale + w * v;
      }
    }
  }

  // ---- wave-level reduction over the 16 edge columns ---------------------
#pragma unroll
  for (int mt = 0; mt < 8; ++mt)
#pragma unroll
    for (int r = 0; r < 8; ++r) {
      float a = acc[mt][r];
      a += __shfl_xor(a, 1, 32);
      a += __shfl_xor(a, 2, 32);
      a += __shfl_xor(a, 4, 32);
      a += __shfl_xor(a, 8, 32);
      acc[mt][r] = a;
    }
  float Z = Zp;
  Z += __shfl_xor(Z, 1, 32);
  Z += __shfl_xor(Z, 2, 32);
  Z += __shfl_xor(Z, 4, 32);
  Z += __shfl_xor(Z, 8, 32);

  float* pp = partials + (size_t)wid * 130;
  if (lane == 0) { pp[0] = mx; pp[1] = Z; }
  if (n == 0) {  // lanes 0 and 16 cover complementary m halves
#pragma unroll
    for (int mt = 0; mt < 8; ++mt)
#pragma unroll
      for (int r = 0; r < 8; ++r)
        pp[2 + mt * 16 + 8 * half + r] = acc[mt][r];
  }
}

// ---------------------------------------------------------------------------
// Kernel 3: merge per-wave softmax partials, then fc4 head (128->256->64).
// ---------------------------------------------------------------------------
__global__ void gnn_combine_kernel(const float* __restrict__ partials, int nW,
                                   const float* __restrict__ W4a,
                                   const float* __restrict__ b4a,
                                   const float* __restrict__ W4b,
                                   const float* __restrict__ b4b,
                                   float* __restrict__ out) {
  __shared__ float red[256];
  __shared__ float wv[128];
  __shared__ float h4[256];
  const int tid = threadIdx.x;  // 256 threads
  const float NEGINF = -__builtin_inff();

  float lm = NEGINF;
  for (int i = tid; i < nW; i += 256)
    lm = fmaxf(lm, partials[(size_t)i * 130]);
  red[tid] = lm;
  __syncthreads();
  for (int s = 128; s > 0; s >>= 1) {
    if (tid < s) red[tid] = fmaxf(red[tid], red[tid + s]);
    __syncthreads();
  }
  const float M = red[0];
  __syncthreads();

  float lz = 0.f;
  if (M != NEGINF)
    for (int i = tid; i < nW; i += 256)
      lz += partials[(size_t)i * 130 + 1] * __expf(partials[(size_t)i * 130] - M);
  red[tid] = lz;
  __syncthreads();
  for (int s = 128; s > 0; s >>= 1) {
    if (tid < s) red[tid] += red[tid + s];
    __syncthreads();
  }
  const float Z = red[0];
  __syncthreads();

  if (tid < 128) {
    float a = 0.f;
    if (Z > 0.f) {
      for (int i = 0; i < nW; ++i) {
        float ei = __expf(partials[(size_t)i * 130] - M);
        a += partials[(size_t)i * 130 + 2 + tid] * ei;
      }
      a /= Z;
    }
    wv[tid] = a;  // weighted[128]; exact 0 when no edges (matches reference)
  }
  __syncthreads();

  {  // h4 = relu(weighted @ W4a + b4a), W4a: [128,256]
    float s = b4a[tid];
    for (int m = 0; m < 128; ++m) s = fmaf(wv[m], W4a[m * 256 + tid], s);
    h4[tid] = fmaxf(s, 0.f);
  }
  __syncthreads();

  if (tid < 64) {  // out = h4 @ W4b + b4b, W4b: [256,64]
    float s = b4b[tid];
    for (int k = 0; k < 256; ++k) s = fmaf(h4[k], W4b[k * 64 + tid], s);
    out[tid] = s;
  }
}

// ---------------------------------------------------------------------------
extern "C" void kernel_launch(void* const* d_in, const int* in_sizes, int n_in,
                              void* d_out, int out_size, void* d_ws,
                              size_t ws_size, hipStream_t stream) {
  // 0 node_obs (unused) 1 adj 2 agent_id (unused)
  // 3 W1a 4 b1a 5 W1b 6 b1b 7 W2a 8 b2a 9 W2b 10 b2b(cancels)
  // 11 W3a 12 b3a 13 W3b 14 b3b 15 W4a 16 b4a 17 W4b 18 b4b
  const float* adj = (const float*)d_in[1];
  const float* W1a = (const float*)d_in[3];
  const float* b1a = (const float*)d_in[4];
  const float* W1b = (const float*)d_in[5];
  const float* b1b = (const float*)d_in[6];
  const float* W2a = (const float*)d_in[7];
  const float* b2a = (const float*)d_in[8];
  const float* W2b = (const float*)d_in[9];
  const float* W3a = (const float*)d_in[11];
  const float* b3a = (const float*)d_in[12];
  const float* W3b = (const float*)d_in[13];
  const float* b3b = (const float*)d_in[14];
  const float* W4a = (const float*)d_in[15];
  const float* b4a = (const float*)d_in[16];
  const float* W4b = (const float*)d_in[17];
  const float* b4b = (const float*)d_in[18];

  const int P = in_sizes[1];           // 8*256*256 = 524288 edge slots
  const int nTiles = (P + 15) / 16;

  _Float16* hw = (_Float16*)d_ws;      // padded weight arena: 239616 B
  float* partials = (float*)((char*)d_ws + 240128);  // 512B-aligned after arena

  const int blocks = 1024, tpb = WPB * 32;
  const int nWaves = blocks * tpb / 32;  // 4096 -> partials = 2.08 MB

  gnn_prep_kernel<<<120, 256, 0, stream>>>(W1b, W2a, W3a, W3b, hw);
  gnn_edge_kernel<<<blocks, tpb, 0, stream>>>(adj, P, nTiles, nWaves, hw, W1a,
                                              b1a, b1b, b2a, W2b, b3a, b3b,
                                              partials);
  gnn_combine_kernel<<<1, 256, 0, stream>>>(partials, nWaves, W4a, b4a, W4b,
                                            b4b, (float*)d_out);
}
